// MultiHeadPixelAttention_73830487818437
// MI455X (gfx1250) — matrix operand order repaired and checked by reading, not compiled, not run on hardware
//
#include <hip/hip_runtime.h>

// ---------------------------------------------------------------------------
// MultiHeadPixelAttention fused pipeline for gfx1250 (MI455X), bf16 WMMA path.
// ---------------------------------------------------------------------------

typedef __attribute__((ext_vector_type(16))) __bf16 v16bf;
typedef __attribute__((ext_vector_type(8)))  __bf16 v8bf;
typedef __attribute__((ext_vector_type(8)))  float  v8f;

#define NPIX   65536   // 4*128*128
#define IMG_H  128
#define IMG_W  128
#define DCH    64

__device__ __forceinline__ unsigned short f2bf(float f) {
    union { float f; unsigned u; } v; v.f = f;
    unsigned r = v.u + 0x7fffu + ((v.u >> 16) & 1u);   // round-to-nearest-even
    return (unsigned short)(r >> 16);
}

// A fragment: 16x32 bf16, row-major source (rows = M, stride in elements).
// Lane<16: M=lane, K=0..15; lane>=16: M=lane-16, K=16..31.
__device__ __forceinline__ v16bf load_a(const unsigned short* base, int rowstride, int lane) {
    int m  = lane & 15;
    int kb = (lane >> 4) ? 16 : 0;
    const unsigned short* p = base + m * rowstride + kb;
    v8bf lo = *(const v8bf*)p;          // K kb..kb+7
    v8bf hi = *(const v8bf*)(p + 8);    // K kb+8..kb+15
    return __builtin_shufflevector(lo, hi, 0,1,2,3,4,5,6,7,8,9,10,11,12,13,14,15);
}

// B fragment: 32x16 bf16 from transposed weights Wt[Nout][K].
// Lane<16: K=k0..k0+15 of column n0+lane; lane>=16: K=k0+16..k0+31.
__device__ __forceinline__ v16bf load_b(const unsigned short* wt, int K, int n0, int k0, int lane) {
    int n  = n0 + (lane & 15);
    int kb = k0 + ((lane >> 4) ? 16 : 0);
    return *(const v16bf*)(wt + (size_t)n * K + kb);
}

__device__ __forceinline__ v8f wmma_bf16(v16bf a, v16bf b, v8f c) {
    return __builtin_amdgcn_wmma_f32_16x16x32_bf16(false, a, false, b, (short)0, c, false, false);
}

// ---------------------------------------------------------------------------
// Prep kernels: dtype conversion + weight transposition to [Nout][K] bf16
// ---------------------------------------------------------------------------
__global__ void f32_to_bf16_kernel(const float* __restrict__ in,
                                   unsigned short* __restrict__ out, int n) {
    for (int i = blockIdx.x * blockDim.x + threadIdx.x; i < n; i += gridDim.x * blockDim.x)
        out[i] = f2bf(in[i]);
}

// out[b][c][r] = bf16(in[b][r][c])
__global__ void transpose_to_bf16_kernel(const float* __restrict__ in,
                                         unsigned short* __restrict__ out,
                                         int batch, int rows, int cols) {
    int total = batch * rows * cols;
    for (int i = blockIdx.x * blockDim.x + threadIdx.x; i < total; i += gridDim.x * blockDim.x) {
        int b = i / (rows * cols);
        int rem = i % (rows * cols);
        int r = rem / cols, c = rem % cols;
        out[((size_t)b * cols + c) * rows + r] = f2bf(in[i]);
    }
}

// ---------------------------------------------------------------------------
// Fused multi-head pixel attention: one 16x16 tile per block, 8 waves.
// ---------------------------------------------------------------------------
__device__ __forceinline__ void tile_gemm64(const unsigned short* __restrict__ xh,
                                            const unsigned short* __restrict__ wt,
                                            float* __restrict__ buf,
                                            int dy, int dx, int wid, int lane) {
    // 16 m-tiles x 4 n-tiles; wave-uniform loop (EXEC all ones for WMMA).
    for (int tile = wid; tile < 64; tile += 8) {
        int mtile = tile & 15;           // = py of the 16 pixels in this tile row
        int n0 = (tile >> 4) * 16;
        v8f c = {};
        const unsigned short* abase = xh + ((mtile + 1 + dy) * 18 + (1 + dx)) * DCH;
        #pragma unroll
        for (int kc = 0; kc < 2; ++kc) {
            v16bf a = load_a(abase + kc * 32, DCH, lane);
            v16bf b = load_b(wt, DCH, n0, kc * 32, lane);
            c = wmma_bf16(a, b, c);
        }
        int n  = n0 + (lane & 15);
        int mb = mtile * 16 + ((lane >> 4) ? 8 : 0);
        #pragma unroll
        for (int r = 0; r < 8; ++r) buf[(mb + r) * DCH + n] = c[r];
    }
}

__global__ __launch_bounds__(256)
void attn_kernel(const unsigned short* __restrict__ xb,
                 const unsigned short* __restrict__ Kt,   // [9][64][64]
                 const unsigned short* __restrict__ Vt,   // [9][64][64]
                 const unsigned short* __restrict__ Qt,   // [64][64]
                 unsigned short* __restrict__ attnb) {
    extern __shared__ char smem[];
    unsigned short* xh = (unsigned short*)smem;                 // 18*18*64 bf16
    float* qbuf = (float*)(smem + 18 * 18 * DCH * 2);           // 256*64 f32
    float* kv   = qbuf + 256 * DCH;                             // 256*64 f32

    const int tileX = blockIdx.x, tileY = blockIdx.y, b = blockIdx.z;
    const int tid = threadIdx.x, wid = tid >> 5, lane = tid & 31;

    // Load x halo tile (zero-padded) -> LDS. Zero halo makes padded taps'
    // projections exactly zero, matching the reference mask semantics.
    for (int idx = tid; idx < 18 * 18 * (DCH / 8); idx += 256) {
        int pix = idx >> 3, ch = idx & 7;
        int hy = pix / 18, hx = pix % 18;
        int gy = tileY * 16 - 1 + hy, gx = tileX * 16 - 1 + hx;
        uint4 v = {0u, 0u, 0u, 0u};
        if ((unsigned)gy < (unsigned)IMG_H && (unsigned)gx < (unsigned)IMG_W)
            v = *(const uint4*)(xb + (((size_t)b * IMG_H + gy) * IMG_W + gx) * DCH + ch * 8);
        *(uint4*)(xh + pix * DCH + ch * 8) = v;
    }
    __syncthreads();

    // Q projection for the tile -> qbuf
    tile_gemm64(xh, Qt, qbuf, 0, 0, wid, lane);
    __syncthreads();

    const int p = tid;                   // one pixel per thread
    float s[2][9];

    // Pass 1: per-tap K projection + per-head dot products
    for (int tap = 0; tap < 9; ++tap) {
        __syncthreads();
        tile_gemm64(xh, Kt + tap * DCH * DCH, kv, tap / 3 - 1, tap % 3 - 1, wid, lane);
        __syncthreads();
        float d0 = 0.f, d1 = 0.f;
        #pragma unroll
        for (int c = 0; c < 32; ++c) {
            d0 += qbuf[p * DCH + c]      * kv[p * DCH + c];
            d1 += qbuf[p * DCH + 32 + c] * kv[p * DCH + 32 + c];
        }
        s[0][tap] = d0 * (1.f / 3.f);
        s[1][tap] = d1 * (1.f / 3.f);
    }

    // Softmax over taps, per head (masked taps are exact zeros, as in ref).
    float al[2][9];
    #pragma unroll
    for (int h = 0; h < 2; ++h) {
        float m = s[h][0];
        #pragma unroll
        for (int t = 1; t < 9; ++t) m = fmaxf(m, s[h][t]);
        float sum = 0.f;
        #pragma unroll
        for (int t = 0; t < 9; ++t) { float e = __expf(s[h][t] - m); al[h][t] = e; sum += e; }
        float inv = 1.f / sum;
        #pragma unroll
        for (int t = 0; t < 9; ++t) al[h][t] *= inv;
    }

    // Pass 2: per-tap V projection + weighted accumulation
    float acc[DCH];
    #pragma unroll
    for (int c = 0; c < DCH; ++c) acc[c] = 0.f;
    for (int tap = 0; tap < 9; ++tap) {
        __syncthreads();
        tile_gemm64(xh, Vt + tap * DCH * DCH, kv, tap / 3 - 1, tap % 3 - 1, wid, lane);
        __syncthreads();
        #pragma unroll
        for (int c = 0; c < DCH; ++c) acc[c] += al[c >> 5][tap] * kv[p * DCH + c];
    }

    int py = p >> 4, px = p & 15;
    size_t o = (((size_t)b * IMG_H + tileY * 16 + py) * IMG_W + tileX * 16 + px) * DCH;
    #pragma unroll
    for (int c = 0; c < DCH; ++c) attnb[o + c] = f2bf(acc[c]);
}

// ---------------------------------------------------------------------------
// 1x1 conv (GEMM) + bias + ReLU -> bf16
// ---------------------------------------------------------------------------
__global__ __launch_bounds__(256)
void gemm_bias_relu_kernel(const unsigned short* __restrict__ A,   // [M][K]
                           const unsigned short* __restrict__ Wt,  // [N][K]
                           const float* __restrict__ bias,
                           unsigned short* __restrict__ outb,      // [M][N]
                           int M, int K, int N) {
    int wid = threadIdx.x >> 5, lane = threadIdx.x & 31;
    int mtile = blockIdx.x * 8 + wid;
    int n0 = blockIdx.y * 16;
    int m0 = mtile * 16;
    if (m0 >= M) return;
    v8f c = {};
    for (int k0 = 0; k0 < K; k0 += 32) {
        __builtin_prefetch(A + (size_t)(m0 + 128) * K + k0, 0, 1);
        v16bf a = load_a(A + (size_t)m0 * K + k0, K, lane);
        v16bf b = load_b(Wt, K, n0, k0, lane);
        c = wmma_bf16(a, b, c);
    }
    int n  = n0 + (lane & 15);
    int mb = m0 + ((lane >> 4) ? 8 : 0);
    float bv = bias[n];
    #pragma unroll
    for (int r = 0; r < 8; ++r) {
        float v = c[r] + bv;
        v = v > 0.f ? v : 0.f;
        outb[(size_t)(mb + r) * N + n] = f2bf(v);
    }
}

// ---------------------------------------------------------------------------
// 3x3 SAME conv + bias + ReLU via 9 shifted WMMA GEMMs over an LDS halo tile.
// ---------------------------------------------------------------------------
template<int CIN, int COUT, bool OUT_F32>
__global__ __launch_bounds__(256)
void conv3x3_bias_relu_kernel(const unsigned short* __restrict__ in,  // [B][H][W][CIN]
                              const unsigned short* __restrict__ Wt,  // [9][COUT][CIN]
                              const float* __restrict__ bias,
                              void* __restrict__ out) {
    extern __shared__ char smem[];
    unsigned short* sh = (unsigned short*)smem;       // 18*18*CIN bf16
    const int tileX = blockIdx.x, tileY = blockIdx.y, b = blockIdx.z;
    const int tid = threadIdx.x, wid = tid >> 5, lane = tid & 31;

    // Halo load (zero-padded SAME boundary)
    constexpr int NCH = CIN / 8;
    for (int idx = tid; idx < 18 * 18 * NCH; idx += 256) {
        int pix = idx / NCH, ch = idx % NCH;
        int hy = pix / 18, hx = pix % 18;
        int gy = tileY * 16 - 1 + hy, gx = tileX * 16 - 1 + hx;
        uint4 v = {0u, 0u, 0u, 0u};
        if ((unsigned)gy < (unsigned)IMG_H && (unsigned)gx < (unsigned)IMG_W)
            v = *(const uint4*)(in + (((size_t)b * IMG_H + gy) * IMG_W + gx) * CIN + ch * 8);
        *(uint4*)(sh + pix * CIN + ch * 8) = v;
    }
    __syncthreads();

    constexpr int NT = COUT / 16;
    for (int tile = wid; tile < 16 * NT; tile += 8) {   // wave-uniform
        int mtile = tile % 16;         // py
        int n0 = (tile / 16) * 16;
        v8f c = {};
        #pragma unroll
        for (int tap = 0; tap < 9; ++tap) {
            int dy = tap / 3 - 1, dx = tap % 3 - 1;
            const unsigned short* abase = sh + ((mtile + 1 + dy) * 18 + (1 + dx)) * CIN;
            const unsigned short* wtap = Wt + (size_t)tap * COUT * CIN;
            #pragma unroll
            for (int kc = 0; kc < CIN / 32; ++kc) {
                v16bf a  = load_a(abase + kc * 32, CIN, lane);
                v16bf bm = load_b(wtap, CIN, n0, kc * 32, lane);
                c = wmma_bf16(a, bm, c);
            }
        }
        int n = n0 + (lane & 15);
        int mbase = (lane >> 4) ? 8 : 0;
        float bv = bias[n];
        #pragma unroll
        for (int r = 0; r < 8; ++r) {
            int px = mbase + r;
            int gy = tileY * 16 + mtile, gx = tileX * 16 + px;
            float v = c[r] + bv;
            v = v > 0.f ? v : 0.f;
            size_t o = (((size_t)b * IMG_H + gy) * IMG_W + gx) * COUT + n;
            if (OUT_F32) ((float*)out)[o] = v;
            else ((unsigned short*)out)[o] = f2bf(v);
        }
    }
}

// ---------------------------------------------------------------------------
// Launch
// ---------------------------------------------------------------------------
extern "C" void kernel_launch(void* const* d_in, const int* in_sizes, int n_in,
                              void* d_out, int out_size, void* d_ws, size_t ws_size,
                              hipStream_t stream) {
    const float* x   = (const float*)d_in[0];
    const float* K_P = (const float*)d_in[1];
    const float* V_P = (const float*)d_in[2];
    const float* Q_P = (const float*)d_in[3];
    const float* f1k = (const float*)d_in[4];
    const float* f1b = (const float*)d_in[5];
    const float* f2k = (const float*)d_in[6];
    const float* f2b = (const float*)d_in[7];
    const float* f3k = (const float*)d_in[8];
    const float* f3b = (const float*)d_in[9];
    float* out = (float*)d_out;

    char* ws = (char*)d_ws;
    size_t off = 0;
    auto alloc = [&](size_t bytes) { char* p = ws + off; off += (bytes + 255) & ~(size_t)255; return p; };
    unsigned short* xb    = (unsigned short*)alloc((size_t)NPIX * DCH * 2);
    unsigned short* attnb = (unsigned short*)alloc((size_t)NPIX * DCH * 2);
    unsigned short* t1b   = (unsigned short*)alloc((size_t)NPIX * DCH * 2);
    unsigned short* t2b   = (unsigned short*)alloc((size_t)NPIX * 128 * 2);
    unsigned short* Kt    = (unsigned short*)alloc(9 * 64 * 64 * 2);
    unsigned short* Vt    = (unsigned short*)alloc(9 * 64 * 64 * 2);
    unsigned short* Qt    = (unsigned short*)alloc(64 * 64 * 2);
    unsigned short* f1t   = (unsigned short*)alloc(64 * 64 * 2);
    unsigned short* f2t   = (unsigned short*)alloc(9 * 128 * 64 * 2);
    unsigned short* f3t   = (unsigned short*)alloc(9 * 64 * 128 * 2);

    // --- prep: convert / transpose to bf16 [Nout][K] ---
    f32_to_bf16_kernel<<<1024, 256, 0, stream>>>(x, xb, NPIX * DCH);
    transpose_to_bf16_kernel<<<144, 256, 0, stream>>>(K_P, Kt, 1, 64, 576);
    transpose_to_bf16_kernel<<<144, 256, 0, stream>>>(V_P, Vt, 1, 64, 576);
    transpose_to_bf16_kernel<<<16,  256, 0, stream>>>(Q_P, Qt, 1, 64, 64);
    transpose_to_bf16_kernel<<<16,  256, 0, stream>>>(f1k, f1t, 1, 64, 64);
    transpose_to_bf16_kernel<<<288, 256, 0, stream>>>(f2k, f2t, 9, 64, 128);
    transpose_to_bf16_kernel<<<288, 256, 0, stream>>>(f3k, f3t, 9, 128, 64);

    // --- fused attention ---
    {
        dim3 grid(IMG_W / 16, IMG_H / 16, 4);
        size_t shmem = 18 * 18 * DCH * 2 + 2 * 256 * DCH * sizeof(float);  // 172544 B
        attn_kernel<<<grid, 256, shmem, stream>>>(xb, Kt, Vt, Qt, attnb);
    }

    // --- ff1: 1x1 GEMM + ReLU ---
    {
        dim3 grid(NPIX / 128, DCH / 16);
        gemm_bias_relu_kernel<<<grid, 256, 0, stream>>>(attnb, f1t, f1b, t1b, NPIX, DCH, DCH);
    }

    // --- ff2: 3x3 conv 64 -> 128 + ReLU ---
    {
        dim3 grid(IMG_W / 16, IMG_H / 16, 4);
        size_t shmem = 18 * 18 * 64 * 2;       // 41472 B
        conv3x3_bias_relu_kernel<64, 128, false><<<grid, 256, shmem, stream>>>(t1b, f2t, f2b, t2b);
    }

    // --- ff3: 3x3 conv 128 -> 64 + ReLU -> f32 output ---
    {
        dim3 grid(IMG_W / 16, IMG_H / 16, 4);
        size_t shmem = 18 * 18 * 128 * 2;      // 82944 B
        conv3x3_bias_relu_kernel<128, 64, true><<<grid, 256, shmem, stream>>>(t2b, f3t, f3b, out);
    }
}
